// MLPDecoder_4844723110630
// MI455X (gfx1250) — compile-verified, matrix operand order checked by reading, not run on hardware
//
#include <hip/hip_runtime.h>

typedef __attribute__((ext_vector_type(16))) _Float16 v16h;
typedef __attribute__((ext_vector_type(8)))  _Float16 v8h;
typedef __attribute__((ext_vector_type(8)))  float    v8f;
typedef __attribute__((ext_vector_type(4)))  float    v4f;

#define EMBED  128
#define HIDDEN 64

// ---------------------------------------------------------------------------
// Pass 1: convert Z (fp32) once into an f16 copy in workspace, stored in
// "fragment order": within every 32-feature group the octets are permuted
// [0-7, 16-23, 8-15, 24-31]. With that layout a lane's entire 16-half WMMA
// A-fragment is one contiguous 32B slice:
//   lanes 0-15  read group*32 + 0  .. +15   (old K {0..7, 16..23})
//   lanes 16-31 read group*32 + 16 .. +31   (old K {8..15, 24..31})
// ---------------------------------------------------------------------------
__global__ __launch_bounds__(256) void convert_z_f16_frag(
    const float* __restrict__ Z, _Float16* __restrict__ Zh, int nOct)
{
    const int oct = blockIdx.x * blockDim.x + threadIdx.x;  // 8 elems per oct
    if (oct >= nOct) return;
    const int q8    = oct & 3;                       // octet within 32-group
    const int srcQ8 = ((q8 & 1) << 1) | (q8 >> 1);   // {0,2,1,3} swap
    const int srcOct = (oct & ~3) | srcQ8;
    v4f a = *(const v4f*)(Z + srcOct * 8);
    v4f b = *(const v4f*)(Z + srcOct * 8 + 4);
    v8h o;
    #pragma unroll
    for (int j = 0; j < 4; ++j) { o[j] = (_Float16)a[j]; o[4 + j] = (_Float16)b[j]; }
    *(v8h*)(Zh + oct * 8) = o;
}

// ---------------------------------------------------------------------------
// Pass 2: one wave computes a [16 edges x 64 hidden] tile via 8 K-steps x
// 4 N-tiles of v_wmma_f32_16x16x32_f16. A-fragments are explicitly
// double-buffered (aCur/aNext) so the loads for step kt+1 are in flight
// while the WMMA quad of step kt executes -> partial s_wait_loadcnt.
// ---------------------------------------------------------------------------
__global__ __launch_bounds__(256) void mlp_decoder_wmma_f16(
    const _Float16* __restrict__ Zh,   // [N_NODES, 128] f16, fragment order
    const int*      __restrict__ edges,// [2, E]
    const float*    __restrict__ W1,   // [256, 64] row-major (k, n)
    const float*    __restrict__ b1,   // [64]
    const float*    __restrict__ W2,   // [64, 1]
    const float*    __restrict__ b2,   // [1]
    float* __restrict__ out,           // [E]
    int E, int numTiles)
{
    // W1 as f16 in WMMA B-fragment layout: [kt][nt][lane][p]
    //   K = kt*32 + 16*(p>=8) + 8*(L>=16) + (p&7),  N = nt*16 + (L&15)
    __shared__ alignas(32) _Float16 sW1[8][4][32][16];

    const int tid = threadIdx.x;
    for (int i = tid; i < 8 * 4 * 32 * 16; i += 256) {
        const int p  = i & 15;
        const int L  = (i >> 4) & 31;
        const int nt = (i >> 9) & 3;
        const int kt = i >> 11;
        const int K = kt * 32 + ((p >= 8) ? 16 : 0) + ((L >= 16) ? 8 : 0) + (p & 7);
        const int N = nt * 16 + (L & 15);
        sW1[kt][nt][L][p] = (_Float16)W1[K * HIDDEN + N];
    }
    __syncthreads();

    const int lane = tid & 31;
    const int m16  = lane & 15;     // N column / edge-row selector
    const int hi   = lane >> 4;     // 0: lanes 0-15, 1: lanes 16-31
    const int hi16 = hi * 16;       // contiguous fragment slice offset

    float b1v[4], w2v[4];
    #pragma unroll
    for (int nt = 0; nt < 4; ++nt) {
        b1v[nt] = b1[nt * 16 + m16];
        w2v[nt] = W2[nt * 16 + m16];
    }
    const float b2v = b2[0];

    const int* srcIdx = edges;
    const int* dstIdx = edges + E;

    const int waveId    = tid >> 5;
    const int gridWaves = gridDim.x * 8;

    for (int tile = blockIdx.x * 8 + waveId; tile < numTiles; tile += gridWaves) {
        const int e0 = tile * 16;
        int e = e0 + m16;
        if (e >= E) e = E - 1;                       // tail clamp, keeps EXEC full
        const _Float16* base0 = Zh + (size_t)srcIdx[e] * EMBED + hi16;
        const _Float16* base1 = Zh + (size_t)dstIdx[e] * EMBED + hi16;

        const int nextTile = tile + gridWaves;
        if (nextTile < numTiles) {
            __builtin_prefetch(&srcIdx[nextTile * 16 + m16], 0, 3);
            __builtin_prefetch(&dstIdx[nextTile * 16 + m16], 0, 3);
        }

        v8f acc[4] = {v8f{}, v8f{}, v8f{}, v8f{}};

        // Software pipeline: fragment for kt+1 loads while kt's WMMAs run.
        v16h aCur = *(const v16h*)(base0);
        #pragma unroll
        for (int kt = 0; kt < 8; ++kt) {
            v16h aNext = aCur;
            if (kt < 7) {
                const int kn = kt + 1;
                const _Float16* rp = (kn < 4) ? base0 : base1;
                aNext = *(const v16h*)(rp + (kn & 3) * 32);
            }
            #pragma unroll
            for (int nt = 0; nt < 4; ++nt) {
                v16h bfrag = *(const v16h*)(&sW1[kt][nt][lane][0]);
                acc[nt] = __builtin_amdgcn_wmma_f32_16x16x32_f16(
                    false, aCur, false, bfrag, (short)0, acc[nt], false, false);
            }
            aCur = aNext;
        }

        // Epilogue: VGPR r holds edge m = r + 8*hi, lane holds n = nt*16+m16.
        #pragma unroll
        for (int r = 0; r < 8; ++r) {
            float s = 0.f;
            #pragma unroll
            for (int nt = 0; nt < 4; ++nt) {
                float h = acc[nt][r] + b1v[nt];
                h = h > 0.f ? h : 0.f;
                s += h * w2v[nt];
            }
            s += __shfl_xor(s, 1, 16);
            s += __shfl_xor(s, 2, 16);
            s += __shfl_xor(s, 4, 16);
            s += __shfl_xor(s, 8, 16);
            const int m = r + hi * 8;
            if (m16 == 0 && (e0 + m) < E)
                out[e0 + m] = s + b2v;
        }
    }
}

// ---------------------------------------------------------------------------
// Fallback (only if ws_size can't hold the f16 Z copy): gather fp32 and
// convert in-kernel. Same WMMA structure.
// ---------------------------------------------------------------------------
__global__ __launch_bounds__(256) void mlp_decoder_wmma_f32(
    const float* __restrict__ Z,
    const int*   __restrict__ edges,
    const float* __restrict__ W1,
    const float* __restrict__ b1,
    const float* __restrict__ W2,
    const float* __restrict__ b2,
    float* __restrict__ out,
    int E, int numTiles)
{
    __shared__ alignas(32) _Float16 sW1[8][4][32][16];

    const int tid = threadIdx.x;
    for (int i = tid; i < 8 * 4 * 32 * 16; i += 256) {
        const int p  = i & 15;
        const int L  = (i >> 4) & 31;
        const int nt = (i >> 9) & 3;
        const int kt = i >> 11;
        const int K = kt * 32 + ((p >= 8) ? 16 : 0) + ((L >= 16) ? 8 : 0) + (p & 7);
        const int N = nt * 16 + (L & 15);
        sW1[kt][nt][L][p] = (_Float16)W1[K * HIDDEN + N];
    }
    __syncthreads();

    const int lane    = tid & 31;
    const int m16     = lane & 15;
    const int hi      = lane >> 4;
    const int halfSel = hi * 8;

    float b1v[4], w2v[4];
    #pragma unroll
    for (int nt = 0; nt < 4; ++nt) {
        b1v[nt] = b1[nt * 16 + m16];
        w2v[nt] = W2[nt * 16 + m16];
    }
    const float b2v = b2[0];

    const int* srcIdx = edges;
    const int* dstIdx = edges + E;

    const int waveId    = tid >> 5;
    const int gridWaves = gridDim.x * 8;

    for (int tile = blockIdx.x * 8 + waveId; tile < numTiles; tile += gridWaves) {
        const int e0 = tile * 16;
        int e = e0 + m16;
        if (e >= E) e = E - 1;
        const float* rowS = Z + (size_t)srcIdx[e] * EMBED;
        const float* rowD = Z + (size_t)dstIdx[e] * EMBED;

        v8f acc[4] = {v8f{}, v8f{}, v8f{}, v8f{}};

        #pragma unroll
        for (int kt = 0; kt < 8; ++kt) {
            const float* row = (kt < 4) ? rowS : rowD;
            const int c0 = ((kt & 3) * 32) + halfSel;
            v4f x0 = *(const v4f*)(row + c0);
            v4f x1 = *(const v4f*)(row + c0 + 4);
            v4f y0 = *(const v4f*)(row + c0 + 16);
            v4f y1 = *(const v4f*)(row + c0 + 20);
            v16h a;
            #pragma unroll
            for (int j = 0; j < 4; ++j) {
                a[j]      = (_Float16)x0[j];
                a[4 + j]  = (_Float16)x1[j];
                a[8 + j]  = (_Float16)y0[j];
                a[12 + j] = (_Float16)y1[j];
            }
            #pragma unroll
            for (int nt = 0; nt < 4; ++nt) {
                v16h bfrag = *(const v16h*)(&sW1[kt][nt][lane][0]);
                acc[nt] = __builtin_amdgcn_wmma_f32_16x16x32_f16(
                    false, a, false, bfrag, (short)0, acc[nt], false, false);
            }
        }

        #pragma unroll
        for (int r = 0; r < 8; ++r) {
            float s = 0.f;
            #pragma unroll
            for (int nt = 0; nt < 4; ++nt) {
                float h = acc[nt][r] + b1v[nt];
                h = h > 0.f ? h : 0.f;
                s += h * w2v[nt];
            }
            s += __shfl_xor(s, 1, 16);
            s += __shfl_xor(s, 2, 16);
            s += __shfl_xor(s, 4, 16);
            s += __shfl_xor(s, 8, 16);
            const int m = r + hi * 8;
            if (m16 == 0 && (e0 + m) < E)
                out[e0 + m] = s + b2v;
        }
    }
}

extern "C" void kernel_launch(void* const* d_in, const int* in_sizes, int n_in,
                              void* d_out, int out_size, void* d_ws, size_t ws_size,
                              hipStream_t stream) {
    const float* Z    = (const float*)d_in[0];
    const int*   edge = (const int*)  d_in[1];
    const float* W1   = (const float*)d_in[2];
    const float* b1   = (const float*)d_in[3];
    const float* W2   = (const float*)d_in[4];
    const float* b2   = (const float*)d_in[5];
    float* out = (float*)d_out;

    const int nZ = in_sizes[0];               // N_NODES * EMBED
    const int E  = in_sizes[1] / 2;           // edge_index is [2, E]
    const int numTiles = (E + 15) / 16;       // 16 edges per wave-tile
    int blocks = (numTiles + 7) / 8;          // 8 waves per 256-thread block
    if (blocks > 4096) blocks = 4096;         // grid-stride amortizes LDS staging

    const size_t zhBytes = (size_t)nZ * sizeof(_Float16);
    // fragment-order layout needs whole 32-element groups (EMBED=128 ensures it)
    if (ws_size >= zhBytes && (nZ & 31) == 0) {
        _Float16* Zh = (_Float16*)d_ws;
        const int nOct = nZ / 8;
        convert_z_f16_frag<<<(nOct + 255) / 256, 256, 0, stream>>>(Z, Zh, nOct);
        mlp_decoder_wmma_f16<<<blocks, 256, 0, stream>>>(Zh, edge, W1, b1, W2, b2,
                                                         out, E, numTiles);
    } else {
        mlp_decoder_wmma_f32<<<blocks, 256, 0, stream>>>(Z, edge, W1, b1, W2, b2,
                                                         out, E, numTiles);
    }
}
